// MoELayer_87969520157158
// MI455X (gfx1250) — matrix operand order repaired and checked by reading, not compiled, not run on hardware
//
#include <hip/hip_runtime.h>

typedef __attribute__((ext_vector_type(16))) __bf16 v16bf;
typedef __attribute__((ext_vector_type(8)))  __bf16 v8bf;
typedef __attribute__((ext_vector_type(8)))  float  v8f;

#define NTOK 4096
#define DM   512
#define NE   8
#define HD   2048
#define CAP  8192          // per-expert list capacity (N*K worst case)
#define EPSF 1e-10f

// ---------------------------------------------------------------------------
// WMMA helper: D = A(16x32 bf16) * B(32x16 bf16) + C(16x16 f32)
// ---------------------------------------------------------------------------
__device__ __forceinline__ v8f wmma_bf16(v16bf a, v16bf b, v8f c) {
  return __builtin_amdgcn_wmma_f32_16x16x32_bf16(
      /*neg_a=*/false, a, /*neg_b=*/false, b,
      /*c_mod=*/(short)0, c, /*reuse_a=*/false, /*reuse_b=*/false);
}

// B-operand K index for half i in lane la (16-bit B, 32x16):
//   K = 8*(la>>4) + i for i < 8; K = 16 + 8*(la>>4) + (i-8) for i >= 8. col = lane & 15.
__device__ __forceinline__ int b_k_of(int la, int i) {
  return ((la >> 4) << 3) + (i < 8 ? i : i + 8);
}

// Load a WMMA A tile for this lane straight from a row-major bf16 row.
// p points at row_base + klo (klo = (lane>=16)?8:0). Halves 0-7 = K klo..klo+7,
// halves 8-15 = K klo+16..klo+23 -- both contiguous 16B runs.
__device__ __forceinline__ v16bf load_a(const __bf16* p) {
  v8bf lo = *(const v8bf*)p;
  v8bf hi = *(const v8bf*)(p + 16);
  v16bf a;
#pragma unroll
  for (int i = 0; i < 8; ++i) { a[i] = lo[i]; a[i + 8] = hi[i]; }
  return a;
}

// Packed bf16 add (VOP3P V_PK_ADD_BF16) on 2 bf16 lanes per dword.
__device__ __forceinline__ unsigned pkadd_bf16(unsigned a, unsigned b) {
  unsigned d;
  asm volatile("v_pk_add_bf16 %0, %1, %2" : "=v"(d) : "v"(a), "v"(b));
  return d;
}

// Same as load_a but summing two sources (the two top-k slots).
__device__ __forceinline__ v16bf load_a_sum(const __bf16* pa, const __bf16* pb) {
  uint4 alo = *(const uint4*)pa;
  uint4 ahi = *(const uint4*)(pa + 16);
  uint4 blo = *(const uint4*)pb;
  uint4 bhi = *(const uint4*)(pb + 16);
  union { unsigned d[8]; v16bf v; } u;
  u.d[0] = pkadd_bf16(alo.x, blo.x);
  u.d[1] = pkadd_bf16(alo.y, blo.y);
  u.d[2] = pkadd_bf16(alo.z, blo.z);
  u.d[3] = pkadd_bf16(alo.w, blo.w);
  u.d[4] = pkadd_bf16(ahi.x, bhi.x);
  u.d[5] = pkadd_bf16(ahi.y, bhi.y);
  u.d[6] = pkadd_bf16(ahi.z, bhi.z);
  u.d[7] = pkadd_bf16(ahi.w, bhi.w);
  return u.v;
}

// ---------------------------------------------------------------------------
// 0) zero counters / expert-prob sums (workspace is poisoned, never memset)
// ---------------------------------------------------------------------------
__global__ void init_kernel(int* __restrict__ cnt, float* __restrict__ esum) {
  int t = threadIdx.x;
  if (t < NE) { cnt[t] = 0; esum[t] = 0.f; }
}

// ---------------------------------------------------------------------------
// 1) x (f32) -> xb (bf16), row-major [NTOK, DM]
// ---------------------------------------------------------------------------
__global__ void convert_x_kernel(const float* __restrict__ x,
                                 __bf16* __restrict__ xb) {
  int base = blockIdx.x * 1024 + threadIdx.x;
#pragma unroll
  for (int u = 0; u < 4; ++u) {
    int i = base + u * 256;
    xb[i] = (__bf16)x[i];
  }
}

// ---------------------------------------------------------------------------
// 2) pack w_gate / w_up into WMMA B-operand register image (bf16)
//    index = ((((e*16 + t)*8 + ns)*16 + kc)*32 + lane)*16 + i
// ---------------------------------------------------------------------------
__global__ void pack_gateup_kernel(const float* __restrict__ wg,
                                   const float* __restrict__ wu,
                                   __bf16* __restrict__ gp,
                                   __bf16* __restrict__ up) {
  int p = blockIdx.x * 256 + threadIdx.x;     // 8,388,608 total
  int i    = p & 15;
  int lane = (p >> 4) & 31;
  int kc   = (p >> 9) & 15;
  int ns   = (p >> 13) & 7;
  int t    = (p >> 16) & 15;
  int e    = p >> 20;
  int col  = e * HD + t * 128 + ns * 16 + (lane & 15);
  int d    = kc * 32 + b_k_of(lane, i);
  size_t off = (size_t)d * (NE * HD) + col;
  gp[p] = (__bf16)wg[off];
  up[p] = (__bf16)wu[off];
}

// ---------------------------------------------------------------------------
// 3) pack w_down [HD, DM]: index = ((ct*64 + kc)*32 + lane)*16 + i
// ---------------------------------------------------------------------------
__global__ void pack_down_kernel(const float* __restrict__ wd,
                                 __bf16* __restrict__ dp) {
  int p = blockIdx.x * 256 + threadIdx.x;     // 1,048,576 total
  int i    = p & 15;
  int lane = (p >> 4) & 31;
  int kc   = (p >> 9) & 63;
  int ct   = p >> 15;
  int col  = ct * 16 + (lane & 15);
  int h    = kc * 32 + b_k_of(lane, i);
  dp[p] = (__bf16)wd[(size_t)h * DM + col];
}

// ---------------------------------------------------------------------------
// 4) router: logits -> softmax -> top-2 -> renorm -> per-expert lists
// ---------------------------------------------------------------------------
__global__ void router_kernel(const float* __restrict__ x,
                              const float* __restrict__ wr,
                              int* __restrict__ listTok,
                              float* __restrict__ listP,
                              int* __restrict__ cnt,
                              float* __restrict__ esum) {
  __shared__ float wS[DM * NE];   // 16 KB router weights
  __shared__ float esS[NE];
  const int tid = threadIdx.x;
  for (int i = tid; i < DM * NE; i += 256) wS[i] = wr[i];
  if (tid < NE) esS[tid] = 0.f;
  __syncthreads();

  const int tok = blockIdx.x * 256 + tid;
  const float* xr = x + (size_t)tok * DM;
  float acc[NE];
#pragma unroll
  for (int e = 0; e < NE; ++e) acc[e] = 0.f;
  for (int d = 0; d < DM; ++d) {
    float xv = xr[d];
#pragma unroll
    for (int e = 0; e < NE; ++e) acc[e] += xv * wS[d * NE + e];
  }
  float mx = acc[0];
#pragma unroll
  for (int e = 1; e < NE; ++e) mx = fmaxf(mx, acc[e]);
  float sum = 0.f;
#pragma unroll
  for (int e = 0; e < NE; ++e) { acc[e] = __expf(acc[e] - mx); sum += acc[e]; }
  float inv = 1.f / sum;
#pragma unroll
  for (int e = 0; e < NE; ++e) {
    acc[e] *= inv;
    atomicAdd(&esS[e], acc[e]);
  }
  int i1 = 0; float p1 = acc[0];
#pragma unroll
  for (int e = 1; e < NE; ++e) if (acc[e] > p1) { p1 = acc[e]; i1 = e; }
  int i2 = (i1 == 0) ? 1 : 0; float p2 = acc[i2];
#pragma unroll
  for (int e = 0; e < NE; ++e)
    if (e != i1 && acc[e] > p2) { p2 = acc[e]; i2 = e; }
  float inv2 = 1.f / (p1 + p2 + EPSF);

  int pos1 = atomicAdd(&cnt[i1], 1);
  listTok[i1 * CAP + pos1] = tok;                 // slot 0
  listP[i1 * CAP + pos1]   = p1 * inv2;
  int pos2 = atomicAdd(&cnt[i2], 1);
  listTok[i2 * CAP + pos2] = tok | (1 << 16);     // slot 1
  listP[i2 * CAP + pos2]   = p2 * inv2;

  __syncthreads();
  if (tid < NE) atomicAdd(&esum[tid], esS[tid]);
}

// ---------------------------------------------------------------------------
// 5) per-expert gate/up GEMM: 32 tokens x 128 H-cols per block, 8 waves.
//    Barrier-free inner loop: each wave gathers its own A tiles directly
//    from global (2x b128 per tile), reuses each B panel for 2 WMMAs.
//    contrib[slot][token][h] = p * silu(x@Wg) * (x@Wu)   (bf16)
// ---------------------------------------------------------------------------
__global__ void expert_gemm_kernel(const __bf16* __restrict__ xb,
                                   const __bf16* __restrict__ gp,
                                   const __bf16* __restrict__ upk,
                                   const int*   __restrict__ listTok,
                                   const float* __restrict__ listP,
                                   const int*   __restrict__ cntArr,
                                   __bf16* __restrict__ contrib) {
  const int e  = blockIdx.x;     // expert
  const int t  = blockIdx.y;     // 128-col tile of H
  const int tt = blockIdx.z;     // 32-token tile of this expert's list
  const int cnt = cntArr[e];
  if (tt * 32 >= cnt) return;

  __shared__ int   tokT[32];
  __shared__ int   slotT[32];
  __shared__ float pT[32];

  const int tid = threadIdx.x;
  if (tid < 32) {
    int idx = tt * 32 + tid;
    if (idx < cnt) {
      int entry  = listTok[e * CAP + idx];
      tokT[tid]  = entry & 0xffff;
      slotT[tid] = entry >> 16;
      pT[tid]    = listP[e * CAP + idx];
    } else { tokT[tid] = 0; slotT[tid] = 0; pT[tid] = 0.f; }
  }
  __syncthreads();

  const int lane = tid & 31;
  const int w    = tid >> 5;                 // wave id: owns 16 cols
  const int r    = lane & 15;
  const int klo  = (lane >> 4) << 3;         // 0 or 8

  const __bf16* a0 = xb + (size_t)tokT[r]      * DM + klo;   // M-subtile 0
  const __bf16* a1 = xb + (size_t)tokT[16 + r] * DM + klo;   // M-subtile 1
  const size_t pbase = (size_t)(((e * 16 + t) * 8 + w) * 16) * 512 + lane * 16;
  const __bf16* gb = gp  + pbase;
  const __bf16* ub = upk + pbase;

  v8f accG0 = {}, accG1 = {}, accU0 = {}, accU1 = {};

#pragma unroll 4
  for (int kc = 0; kc < 16; ++kc) {
    v16bf A0 = load_a(a0 + kc * 32);
    v16bf A1 = load_a(a1 + kc * 32);
    v16bf Bg = *(const v16bf*)(gb + (size_t)kc * 512);
    v16bf Bu = *(const v16bf*)(ub + (size_t)kc * 512);
    if (kc + 1 < 16) {
      __builtin_prefetch(gb + (size_t)(kc + 1) * 512, 0, 3);
      __builtin_prefetch(ub + (size_t)(kc + 1) * 512, 0, 3);
    }
    accG0 = wmma_bf16(A0, Bg, accG0);
    accG1 = wmma_bf16(A1, Bg, accG1);
    accU0 = wmma_bf16(A0, Bu, accU0);
    accU1 = wmma_bf16(A1, Bu, accU1);
  }

  // epilogue: silu(g)*u*p -> contrib[slot][token][col]
  const int nn = lane & 15;
  const int rb = (lane >> 4) << 3;
  const int colBase = t * 128 + w * 16;
#pragma unroll
  for (int j = 0; j < 8; ++j) {
    int row0 = rb + j;            // subtile 0 rows 0..15
    int row1 = 16 + rb + j;       // subtile 1 rows 16..31
    if (tt * 32 + row0 < cnt) {
      float g = accG0[j], u = accU0[j];
      float s = g / (1.f + __expf(-g));
      float val = pT[row0] * s * u;
      size_t off = (size_t)slotT[row0] * (size_t)(NTOK * HD)
                 + (size_t)tokT[row0] * HD + colBase + nn;
      contrib[off] = (__bf16)val;
    }
    if (tt * 32 + row1 < cnt) {
      float g = accG1[j], u = accU1[j];
      float s = g / (1.f + __expf(-g));
      float val = pT[row1] * s * u;
      size_t off = (size_t)slotT[row1] * (size_t)(NTOK * HD)
                 + (size_t)tokT[row1] * HD + colBase + nn;
      contrib[off] = (__bf16)val;
    }
  }
}

// ---------------------------------------------------------------------------
// 6) down projection: out = (c0 + c1) @ w_down, f32 output.
//    Barrier-free, LDS-free: slot sum done with v_pk_add_bf16 in-register.
// ---------------------------------------------------------------------------
__global__ void down_gemm_kernel(const __bf16* __restrict__ contrib,
                                 const __bf16* __restrict__ dp,
                                 float* __restrict__ out) {
  const int tt = blockIdx.x;     // 32-token tile
  const int cg = blockIdx.y;     // 128-col group of DM
  const int tid = threadIdx.x, lane = tid & 31, w = tid >> 5;
  const int tok0 = tt * 32;
  const int ct = cg * 8 + w;     // 16-col tile index
  const int r = lane & 15;
  const int klo = (lane >> 4) << 3;

  const __bf16* c0 = contrib;
  const __bf16* c1 = contrib + (size_t)NTOK * HD;
  const __bf16* p00 = c0 + (size_t)(tok0 + r)      * HD + klo;
  const __bf16* p01 = c1 + (size_t)(tok0 + r)      * HD + klo;
  const __bf16* p10 = c0 + (size_t)(tok0 + 16 + r) * HD + klo;
  const __bf16* p11 = c1 + (size_t)(tok0 + 16 + r) * HD + klo;
  const __bf16* bbase = dp + (size_t)(ct * 64) * 512 + lane * 16;

  v8f acc0 = {}, acc1 = {};

#pragma unroll 4
  for (int kc = 0; kc < 64; ++kc) {
    v16bf A0 = load_a_sum(p00 + kc * 32, p01 + kc * 32);
    v16bf A1 = load_a_sum(p10 + kc * 32, p11 + kc * 32);
    v16bf B  = *(const v16bf*)(bbase + (size_t)kc * 512);
    if (kc + 1 < 64)
      __builtin_prefetch(bbase + (size_t)(kc + 1) * 512, 0, 3);
    acc0 = wmma_bf16(A0, B, acc0);
    acc1 = wmma_bf16(A1, B, acc1);
  }

  const int nn = lane & 15, rb = (lane >> 4) << 3;
  const int colBase = ct * 16;
#pragma unroll
  for (int j = 0; j < 8; ++j) {
    out[(size_t)(tok0 + rb + j)      * DM + colBase + nn] = acc0[j];
    out[(size_t)(tok0 + 16 + rb + j) * DM + colBase + nn] = acc1[j];
  }
}

// ---------------------------------------------------------------------------
// 7) load-balancing aux loss
// ---------------------------------------------------------------------------
__global__ void lb_loss_kernel(const float* __restrict__ esum,
                               float* __restrict__ out) {
  if (threadIdx.x == 0 && blockIdx.x == 0) {
    float lb = 0.f;
    for (int e = 0; e < NE; ++e) {
      float m = esum[e] / (float)NTOK;
      lb += m * __logf(m * (float)NE + EPSF);
    }
    out[(size_t)NTOK * DM] = (float)NE * lb;
  }
}

// ---------------------------------------------------------------------------
// launcher
// ---------------------------------------------------------------------------
extern "C" void kernel_launch(void* const* d_in, const int* in_sizes, int n_in,
                              void* d_out, int out_size, void* d_ws, size_t ws_size,
                              hipStream_t stream) {
  const float* x  = (const float*)d_in[0];
  const float* wr = (const float*)d_in[1];
  const float* wg = (const float*)d_in[2];
  const float* wu = (const float*)d_in[3];
  const float* wd = (const float*)d_in[4];
  float* out = (float*)d_out;

  char* ws = (char*)d_ws;
  constexpr size_t SZ_XB = (size_t)NTOK * DM * 2;                  //  4 MiB
  constexpr size_t SZ_GP = (size_t)NE * 16 * 8 * 16 * 32 * 16 * 2; // 16 MiB
  constexpr size_t SZ_DP = (size_t)32 * 64 * 32 * 16 * 2;          //  2 MiB
  constexpr size_t SZ_CT = (size_t)2 * NTOK * HD * 2;              // 32 MiB
  constexpr size_t SZ_LT = (size_t)NE * CAP * 4;

  size_t o = 0;
  __bf16* xb   = (__bf16*)(ws + o); o += SZ_XB;
  __bf16* gp   = (__bf16*)(ws + o); o += SZ_GP;
  __bf16* up   = (__bf16*)(ws + o); o += SZ_GP;
  __bf16* dp   = (__bf16*)(ws + o); o += SZ_DP;
  __bf16* ct   = (__bf16*)(ws + o); o += SZ_CT;
  int*   listTok = (int*)  (ws + o); o += SZ_LT;
  float* listP   = (float*)(ws + o); o += SZ_LT;
  int*   cnt     = (int*)  (ws + o); o += 64;
  float* esum    = (float*)(ws + o); o += 64;
  (void)ws_size; (void)in_sizes; (void)n_in; (void)out_size;

  init_kernel<<<1, 32, 0, stream>>>(cnt, esum);
  convert_x_kernel<<<2048, 256, 0, stream>>>(x, xb);
  pack_gateup_kernel<<<32768, 256, 0, stream>>>(wg, wu, gp, up);
  pack_down_kernel<<<4096, 256, 0, stream>>>(wd, dp);
  router_kernel<<<NTOK / 256, 256, 0, stream>>>(x, wr, listTok, listP, cnt, esum);
  expert_gemm_kernel<<<dim3(NE, 16, 256), 256, 0, stream>>>(
      xb, gp, up, listTok, listP, cnt, ct);
  down_gemm_kernel<<<dim3(NTOK / 32, 4), 256, 0, stream>>>(ct, dp, out);
  lb_loss_kernel<<<1, 32, 0, stream>>>(esum, out);
}
